// ComplexScaledDotProductAttention_37306085933314
// MI455X (gfx1250) — matrix operand order repaired and checked by reading, not compiled, not run on hardware
//
#include <hip/hip_runtime.h>
#include <hip/hip_bf16.h>

typedef __attribute__((ext_vector_type(16))) _Float16     v16h;
typedef __attribute__((ext_vector_type(8)))  _Float16     v8h;
typedef __attribute__((ext_vector_type(8)))  float        v8f;
typedef __attribute__((ext_vector_type(4)))  unsigned int u32x4;
typedef __attribute__((ext_vector_type(8)))  int          i32x8;
typedef __attribute__((ext_vector_type(4)))  int          i32x4;

#define B_  2
#define H_  12
#define S_  2048
#define D_  64

constexpr int ROWS   = 16;    // query rows per workgroup
constexpr int SCST   = 2052;  // score LDS row stride (floats): 16B-aligned rows, de-aliased banks
constexpr int KST    = 68;    // K LDS row stride (floats): TDM pad 4 dw per 64 dw -> aligned, de-aliased
constexpr int VTST   = 40;    // Vt LDS row stride (halves): 80B rows -> 16B aligned
constexpr int VTROWS = 64;    // d dimension
constexpr int CHUNK  = 32;    // keys per P@V chunk (one WMMA K step)
constexpr int NCHUNK = S_ / CHUNK;
constexpr int KTILES = 256 / 16;                 // 16 key tiles per wave strip
constexpr int KBUF_W = 2 * 2 * 16 * KST;         // per-wave K stage floats (2 buf x {r,i} x 16 x KST)

__device__ __forceinline__ v8f wmma16(v16h a, v16h b, v8f c) {
  return __builtin_amdgcn_wmma_f32_16x16x32_f16(false, a, false, b, (short)0, c, false, false);
}

// A-operand (16x32 f16) from row-major fp32, leading dim ld.
// A layout: lane<16 -> row=lane, k=0..15; lane>=16 -> row=lane-16, k=16..31.
__device__ __forceinline__ v16h load_a16x32(const float* p, int ld, int lane, float scale) {
  const int r = lane & 15, s = lane >> 4;
  const float* a = p + (size_t)r * ld + s * 16;
  float4 x0 = *(const float4*)(a);
  float4 x1 = *(const float4*)(a + 4);
  float4 y0 = *(const float4*)(a + 8);
  float4 y1 = *(const float4*)(a + 12);
  float f[16] = {x0.x, x0.y, x0.z, x0.w, x1.x, x1.y, x1.z, x1.w,
                 y0.x, y0.y, y0.z, y0.w, y1.x, y1.y, y1.z, y1.w};
  v16h o;
#pragma unroll
  for (int j = 0; j < 16; ++j) o[j] = (_Float16)(f[j] * scale);
  return o;
}

// B-operand (32x16 f16) for A@M^T from row-major fp32 M (row = "n", k contiguous).
// B layout used here (contiguous k, same as A): lane<16 -> n=lane, k=0..15; lane>=16 -> n=lane-16, k=16..31.
__device__ __forceinline__ v16h load_b16x32(const float* p, int ld, int lane) {
  const int n = lane & 15, s = lane >> 4;
  const float* a = p + (size_t)n * ld + s * 16;
  float4 x0 = *(const float4*)(a);
  float4 x1 = *(const float4*)(a + 4);
  float4 x2 = *(const float4*)(a + 8);
  float4 x3 = *(const float4*)(a + 12);
  float f[16] = {x0.x, x0.y, x0.z, x0.w, x1.x, x1.y, x1.z, x1.w,
                 x2.x, x2.y, x2.z, x2.w, x3.x, x3.y, x3.z, x3.w};
  v16h o;
#pragma unroll
  for (int j = 0; j < 16; ++j) o[j] = (_Float16)f[j];
  return o;
}

// B-operand (32x16) from the transposed f16 V tile in LDS: Vt[d][key], stride VTST halves.
__device__ __forceinline__ v16h load_b_vt(const _Float16* vt, int d0, int lane) {
  const int n = lane & 15, s = lane >> 4;
  const _Float16* p = vt + (size_t)(d0 + n) * VTST + s * 16;
  v8h lo = *(const v8h*)(p);
  v8h hi = *(const v8h*)(p + 8);
  v16h o;
#pragma unroll
  for (int j = 0; j < 8; ++j) { o[j] = lo[j]; o[8 + j] = hi[j]; }
  return o;
}

// Issue one TDM 1D tile copy: 1024 fp32 elements global -> LDS, LDS padded 4 DWORDs
// every 64 DWORDs (rows land at stride KST=68 floats).  D# per CDNA5 ISA.
__device__ __forceinline__ void tdm_issue(const float* gptr, unsigned lds_addr) {
  const unsigned long long ga = (unsigned long long)(size_t)gptr;
  u32x4 g0;
  g0[0] = 1u;                                               // count=1, user desc
  g0[1] = lds_addr;                                         // lds_addr (bytes)
  g0[2] = (unsigned)(ga & 0xffffffffu);                     // global_addr[31:0]
  g0[3] = (unsigned)((ga >> 32) & 0x01ffffffu) | (2u << 30);// global_addr[56:32], type=2
  i32x8 g1;
  g1[0] = (int)((2u << 16)      // data_size = 4B
              | (1u << 20)      // pad_enable
              | (5u << 22)      // pad_interval: 64 DWORDs
              | (3u << 25));    // pad_amount: 4 DWORDs
  g1[1] = (int)(1024u << 16);   // tensor_dim0 = 1024 (lo16)
  g1[2] = (int)(1u << 16);      // tensor_dim0 hi=0 ; tensor_dim1 = 1
  g1[3] = (int)(1024u << 16);   // tensor_dim1 hi=0 ; tile_dim0 = 1024
  g1[4] = 0;                    // tile_dim1 = 0 (unused), tile_dim2 = 0
  g1[5] = 1024;                 // tensor_dim0_stride lo32
  g1[6] = 0;
  g1[7] = 0;
  const i32x4 z4 = {0, 0, 0, 0};
  const i32x8 z8 = {0, 0, 0, 0, 0, 0, 0, 0};
  // 6-arg variant on this toolchain: (g0, g1, g2, g3, <extra g>, cpol)
  __builtin_amdgcn_tensor_load_to_lds(g0, g1, z4, z4, z8, 0);
}

__global__ __launch_bounds__(256, 1)
void cattn_kernel(const float* __restrict__ q_real, const float* __restrict__ q_imag,
                  const float* __restrict__ k_real, const float* __restrict__ k_imag,
                  const float* __restrict__ v_real, const float* __restrict__ v_imag,
                  const int*   __restrict__ mask,
                  float* __restrict__ out_r, float* __restrict__ out_i,
                  float* __restrict__ attn) {
  extern __shared__ __align__(16) char smem_raw[];
  float*    sc     = (float*)smem_raw;                                   // [ROWS][SCST]
  float*    kstage = sc + (size_t)ROWS * SCST;                           // [8 waves][2 buf][r,i][16][KST]
  _Float16* vt     = (_Float16*)(kstage + 8 * KBUF_W);                   // [2 buf][r,i][64][VTST]
  float*    rowinv = (float*)(vt + 2 * 2 * VTROWS * VTST);               // [ROWS]

  const int tid  = threadIdx.x;
  const int lane = tid & 31;
  const int wave = tid >> 5;                  // 8 waves: column strip in phase 1

  const int bh = blockIdx.x >> 7;             // S_/ROWS == 128 row blocks per head
  const int rb = blockIdx.x & 127;
  const int r0 = rb * ROWS;
  const int b  = bh / H_;

  const size_t head = (size_t)bh * S_ * D_;

  // ---- Phase 1: masked |complex| scores, K tiles streamed in by TDM -------
  const size_t qoff = head + (size_t)r0 * D_;
  const float inv_t = 0.125f;  // 1/TEMP folded into Q
  v16h qr0 = load_a16x32(q_real + qoff,      D_, lane, inv_t);
  v16h qr1 = load_a16x32(q_real + qoff + 32, D_, lane, inv_t);
  v16h qi0 = load_a16x32(q_imag + qoff,      D_, lane, inv_t);
  v16h qi1 = load_a16x32(q_imag + qoff + 32, D_, lane, inv_t);

  const int* mblk = mask + (size_t)b * S_ * S_;
  float* kb = kstage + wave * KBUF_W;         // this wave's private K double buffer

  auto tdm_load_k = [&](int t, int buf) {     // DMA K tile t -> buffer buf (wave-private)
    const int c0 = wave * 256 + t * 16;
    const size_t goff = head + (size_t)c0 * D_;
    float* dst = kb + buf * (2 * 16 * KST);
    tdm_issue(k_real + goff, (unsigned)(size_t)dst);
    tdm_issue(k_imag + goff, (unsigned)(size_t)(dst + 16 * KST));
  };

  auto score_tile = [&](int t, int buf) {
    const int c0 = wave * 256 + t * 16;
    const float* kbr = kb + buf * (2 * 16 * KST);
    const float* kbi = kbr + 16 * KST;
    v16h kr0 = load_b16x32(kbr,      KST, lane);
    v16h kr1 = load_b16x32(kbr + 32, KST, lane);
    v16h ki0 = load_b16x32(kbi,      KST, lane);
    v16h ki1 = load_b16x32(kbi + 32, KST, lane);

    v8f ar = {}; v8f t1 = {}; v8f t2 = {};
    ar = wmma16(qi0, ki0, ar); ar = wmma16(qi1, ki1, ar);
    ar = wmma16(qr0, kr0, ar); ar = wmma16(qr1, kr1, ar);  // qr.kr + qi.ki
    t1 = wmma16(qi0, kr0, t1); t1 = wmma16(qi1, kr1, t1);  // qi.kr
    t2 = wmma16(qr0, ki0, t2); t2 = wmma16(qr1, ki1, t2);  // qr.ki

    const int sel = lane >> 4, nn = lane & 15;
#pragma unroll
    for (int j = 0; j < 8; ++j) {
      const int rl = j + 8 * sel;                      // C layout: M = j + 8*(lane>=16)
      const float ai = t1[j] - t2[j];
      float sv = sqrtf(ar[j] * ar[j] + ai * ai);
      const int mv = mblk[(size_t)(r0 + rl) * S_ + c0 + nn];
      sv = (mv == 0) ? -1e9f : sv;
      sc[rl * SCST + c0 + nn] = sv;
    }
  };

  tdm_load_k(0, 0);                                    // prologue
  for (int t = 0; t < KTILES - 1; ++t) {
    tdm_load_k(t + 1, (t + 1) & 1);                    // prefetch next tile
    __builtin_amdgcn_s_wait_tensorcnt(2);              // tile t landed (in-order per wave)
    score_tile(t, t & 1);
  }
  __builtin_amdgcn_s_wait_tensorcnt(0);
  score_tile(KTILES - 1, (KTILES - 1) & 1);
  __syncthreads();

  // ---- Phase 2: softmax stats (16 threads per row, shuffle reduce) --------
  {
    const int r = tid >> 4, g = tid & 15;
    float m = -3.0e38f;
    for (int c = g; c < S_; c += 16) m = fmaxf(m, sc[r * SCST + c]);
    m = fmaxf(m, __shfl_xor(m, 1, 16));
    m = fmaxf(m, __shfl_xor(m, 2, 16));
    m = fmaxf(m, __shfl_xor(m, 4, 16));
    m = fmaxf(m, __shfl_xor(m, 8, 16));
    float ssum = 0.f;
    for (int c = g; c < S_; c += 16) {
      const float e = __expf(sc[r * SCST + c] - m);
      sc[r * SCST + c] = e;                            // keep un-normalized exp
      ssum += e;
    }
    ssum += __shfl_xor(ssum, 1, 16);
    ssum += __shfl_xor(ssum, 2, 16);
    ssum += __shfl_xor(ssum, 4, 16);
    ssum += __shfl_xor(ssum, 8, 16);
    if (g == 0) rowinv[r] = 1.0f / ssum;
  }
  __syncthreads();

  // ---- Phase 3: write normalized attn (coalesced), stage first V chunk ----
  auto stage = [&](int chunk, int buf) {
    _Float16* vtr = vt + (size_t)buf * (2 * VTROWS * VTST);
    _Float16* vti = vtr + VTROWS * VTST;
    const float* vr = v_real + head + (size_t)chunk * CHUNK * D_;
    const float* vi = v_imag + head + (size_t)chunk * CHUNK * D_;
    for (int e = tid; e < CHUNK * D_; e += 256) {
      const int key = e >> 6, d = e & 63;              // coalesced read, transposed store
      vtr[d * VTST + key] = (_Float16)vr[e];
      vti[d * VTST + key] = (_Float16)vi[e];
    }
  };
  stage(0, 0);

  float* attn_blk = attn + (size_t)bh * S_ * S_ + (size_t)r0 * S_;
  for (int idx = tid; idx < ROWS * S_; idx += 256) {
    const int rr = idx >> 11, cc = idx & (S_ - 1);
    attn_blk[(size_t)rr * S_ + cc] = sc[rr * SCST + cc] * rowinv[rr];
  }
  __syncthreads();

  // ---- Phase 4: O = P @ V; wave -> (ri = re/im, dq = 16-wide d tile) ------
  const int dq = wave & 3;
  const int ri = wave >> 2;
  v8f acc = {};
  for (int c = 0; c < NCHUNK; ++c) {
    const int buf = c & 1;
    if (c + 1 < NCHUNK) stage(c + 1, buf ^ 1);         // double-buffered V staging
    v16h pA = load_a16x32(sc + c * CHUNK, SCST, lane, 1.0f);
    const _Float16* vtb = vt + (size_t)buf * (2 * VTROWS * VTST) + ri * (VTROWS * VTST);
    v16h bv = load_b_vt(vtb, dq * 16, lane);
    acc = wmma16(pA, bv, acc);
    __syncthreads();
  }

  float* outp = ri ? out_i : out_r;
  const int sel = lane >> 4, nn = lane & 15;
#pragma unroll
  for (int j = 0; j < 8; ++j) {
    const int rl = j + 8 * sel;
    const size_t o = head + (size_t)(r0 + rl) * D_ + dq * 16 + nn;
    outp[o] = acc[j] * rowinv[rl];
  }
}

extern "C" void kernel_launch(void* const* d_in, const int* in_sizes, int n_in,
                              void* d_out, int out_size, void* d_ws, size_t ws_size,
                              hipStream_t stream) {
  (void)in_sizes; (void)n_in; (void)out_size; (void)d_ws; (void)ws_size;
  const float* q_real = (const float*)d_in[0];
  const float* q_imag = (const float*)d_in[1];
  const float* k_real = (const float*)d_in[2];
  const float* k_imag = (const float*)d_in[3];
  const float* v_real = (const float*)d_in[4];
  const float* v_imag = (const float*)d_in[5];
  const int*   mask   = (const int*)d_in[6];

  float* out   = (float*)d_out;
  const size_t n_od = (size_t)B_ * H_ * S_ * D_;
  float* out_r = out;
  float* out_i = out + n_od;
  float* attn  = out + 2 * n_od;

  const size_t smem = (size_t)ROWS * SCST * sizeof(float)                 // score strip
                    + (size_t)8 * KBUF_W * sizeof(float)                  // per-wave TDM K buffers
                    + (size_t)2 * 2 * VTROWS * VTST * sizeof(_Float16)    // V^T double buffer
                    + (size_t)ROWS * sizeof(float);                       // rowinv
  (void)hipFuncSetAttribute((const void*)cattn_kernel,
                            hipFuncAttributeMaxDynamicSharedMemorySize, (int)smem);

  dim3 grid(B_ * H_ * (S_ / ROWS));  // 24 heads * 128 row blocks = 3072 workgroups
  cattn_kernel<<<grid, 256, smem, stream>>>(q_real, q_imag, k_real, k_imag,
                                            v_real, v_imag, mask,
                                            out_r, out_i, attn);
}
